// TransformerLayer_45758581571717
// MI455X (gfx1250) — matrix operand order repaired and checked by reading, not compiled, not run on hardware
//
#include <hip/hip_runtime.h>
#include <hip/hip_bf16.h>

// ---------------------------------------------------------------------------
// Transformer layer for gfx1250 (MI455X).
// All GEMMs + attention einsums via v_wmma_f32_16x16x32_f16 (wave32 WMMA,
// f32 accumulate). Operands are converted to f16 ONCE (weights in a prep
// pass, activations emitted in f16 by their producer kernels), so GEMM inner
// loops move raw f16 bits global->LDS via the gfx1250 async-to-LDS engine
// with zero conversion VALU in the hot loop.
// ---------------------------------------------------------------------------

typedef __attribute__((ext_vector_type(16))) _Float16 v16h;
typedef __attribute__((ext_vector_type(8)))  float    v8f;
typedef int tl_v4i __attribute__((vector_size(16)));   // matches builtin param

#define TL_D    1024
#define TL_H    16
#define TL_HD   64
#define TL_DFF  4096
#define TL_B    2
#define TL_L    2048
#define TL_M    (TL_B * TL_L)   // 4096 rows
#define TL_EPS  1e-5f

#if __has_builtin(__builtin_amdgcn_global_load_async_to_lds_b128) && \
    __has_builtin(__builtin_amdgcn_s_wait_asynccnt)
#define TL_HAS_ASYNC_LDS 1
#else
#define TL_HAS_ASYNC_LDS 0
#endif

__device__ __forceinline__ v8f vzero8() {
    v8f z;
#pragma unroll
    for (int i = 0; i < 8; ++i) z[i] = 0.0f;
    return z;
}

// Copy 16 bytes (8 f16) global -> LDS. Uses the CDNA5 async-to-LDS engine
// (ASYNCcnt-tracked, no VGPR round trip) when available.
__device__ __forceinline__ void copy16_g2l(const _Float16* __restrict__ g,
                                           _Float16* __restrict__ l) {
#if TL_HAS_ASYNC_LDS
    __builtin_amdgcn_global_load_async_to_lds_b128(
        (__attribute__((address_space(1))) tl_v4i*)g,
        (__attribute__((address_space(3))) tl_v4i*)l, 0, 0);
#else
    *(uint4*)l = *(const uint4*)g;
#endif
}

__device__ __forceinline__ void copy_fence_lds() {
#if TL_HAS_ASYNC_LDS
    __builtin_amdgcn_s_wait_asynccnt(0);
#endif
}

// A-fragment (16x32 f16, MxK): lane<16 -> row M=lane, K in {0..7, 16..23};
// lane>=16 -> row M=lane-16, K in {8..15, 24..31}. `base` points at row, k0.
__device__ __forceinline__ v16h load_a_frag(const _Float16* base, int laneHalf) {
    const _Float16* p0 = base + (laneHalf ? 8 : 0);
    const _Float16* p1 = base + (laneHalf ? 24 : 16);
    v16h r;
#pragma unroll
    for (int i = 0; i < 8; ++i) { r[i] = p0[i]; r[i + 8] = p1[i]; }
    return r;
}

// B-fragment (32x16 f16, KxN), sourced from a row-major [N][K] tile:
// lane<16 -> col N=lane, K in {0..7, 16..23}; lane>=16 -> col N=lane-16, K in {8..15, 24..31}.
__device__ __forceinline__ v16h load_b_frag(const _Float16* rowptr, int laneHalf) {
    const _Float16* p = rowptr + (laneHalf ? 8 : 0);
    v16h r;
#pragma unroll
    for (int i = 0; i < 8; ++i) { r[i] = p[i]; r[i + 8] = p[i + 16]; }
    return r;
}

__device__ __forceinline__ v8f wmma16(v16h a, v16h b, v8f c) {
    return __builtin_amdgcn_wmma_f32_16x16x32_f16(
        /*neg_a=*/false, a, /*neg_b=*/false, b,
        /*c_mod=*/(short)0, c, /*reuse_a=*/false, /*reuse_b=*/false);
}

__device__ __forceinline__ float gelu_exact(float v) {
    return 0.5f * v * (1.0f + erff(v * 0.70710678118654752f));
}

// ---------------------------------------------------------------------------
// fp32 -> f16 bulk convert (weights prep). n4 = elements/4.
// ---------------------------------------------------------------------------
__global__ void tl_cvt_f16_kernel(const float* __restrict__ src,
                                  _Float16* __restrict__ dst, int n4) {
    const int i = blockIdx.x * 256 + threadIdx.x;
    if (i >= n4) return;
    const float4 v = ((const float4*)src)[i];
    _Float16* d = dst + (size_t)i * 4;
    d[0] = (_Float16)v.x; d[1] = (_Float16)v.y;
    d[2] = (_Float16)v.z; d[3] = (_Float16)v.w;
}

// ---------------------------------------------------------------------------
// LayerNorm: one row (D=1024) per block of 256 threads; emits f16 directly.
// ---------------------------------------------------------------------------
__global__ void tl_layernorm_kernel(const float* __restrict__ x,
                                    const float* __restrict__ g,
                                    const float* __restrict__ b,
                                    _Float16* __restrict__ y) {
    __shared__ float s_sum[256];
    __shared__ float s_sq[256];
    const int row = blockIdx.x;
    const int tid = threadIdx.x;
    const float* xr = x + (size_t)row * TL_D;

    float lsum = 0.0f, lsq = 0.0f;
    float vals[4];
#pragma unroll
    for (int i = 0; i < 4; ++i) {
        float v = xr[tid + i * 256];
        vals[i] = v;
        lsum += v;
        lsq  += v * v;
    }
    s_sum[tid] = lsum;
    s_sq[tid]  = lsq;
    __syncthreads();
    for (int off = 128; off > 0; off >>= 1) {
        if (tid < off) {
            s_sum[tid] += s_sum[tid + off];
            s_sq[tid]  += s_sq[tid + off];
        }
        __syncthreads();
    }
    const float mu  = s_sum[0] * (1.0f / TL_D);
    const float var = s_sq[0] * (1.0f / TL_D) - mu * mu;
    const float inv = rsqrtf(var + TL_EPS);
    _Float16* yr = y + (size_t)row * TL_D;
#pragma unroll
    for (int i = 0; i < 4; ++i) {
        const int c = tid + i * 256;
        yr[c] = (_Float16)((vals[i] - mu) * inv * g[c] + b[c]);
    }
}

// ---------------------------------------------------------------------------
// f16 WMMA GEMM: C = act(A[M,K] @ W[N,K]^T + bias) (+ residual).
// Block tile 128x128, BK=64 (2 wmma k-steps per stage), 8 waves (2x4),
// wave tile 64x32 -> 4x2 accumulators -> 16 WMMAs per barrier pair.
// Optional f32 output (with residual) and/or f16 output (to feed next GEMM).
// ---------------------------------------------------------------------------
#define GS 72   // padded LDS row stride in f16 (64 + 8; 144B, 16B aligned)

__global__ __launch_bounds__(256)
void tl_gemm_f16_kernel(const _Float16* __restrict__ A,
                        const _Float16* __restrict__ W,
                        const float* __restrict__ bias,
                        const float* __restrict__ residual,
                        float* __restrict__ C32,
                        _Float16* __restrict__ C16,
                        int M, int N, int K, int act) {
    __shared__ alignas(16) _Float16 As[128 * GS];
    __shared__ alignas(16) _Float16 Bs[128 * GS];

    const int tid  = threadIdx.x;
    const int lane = tid & 31;
    const int wave = tid >> 5;
    const int laneHalf = (lane >> 4) & 1;
    const int lane15   = lane & 15;
    const int waveM = wave >> 2;   // 0..1
    const int waveN = wave & 3;    // 0..3
    const int bm = blockIdx.y * 128;
    const int bn = blockIdx.x * 128;

    v8f acc[4][2];
#pragma unroll
    for (int i = 0; i < 4; ++i)
#pragma unroll
        for (int j = 0; j < 2; ++j) acc[i][j] = vzero8();

    const int ksteps = K >> 6;           // BK = 64
    for (int kt = 0; kt < ksteps; ++kt) {
        const int k0 = kt << 6;
        // Stage tiles: 128 rows x 8 chunks x 8 f16 per matrix; 4 chunks/thread.
#pragma unroll
        for (int p = 0; p < 4; ++p) {
            const int c   = tid + p * 256;   // 0..1023
            const int row = c >> 3;          // 0..127
            const int cg  = c & 7;           // 16B chunk along K
            copy16_g2l(A + (size_t)(bm + row) * K + k0 + cg * 8,
                       &As[row * GS + cg * 8]);
            copy16_g2l(W + (size_t)(bn + row) * K + k0 + cg * 8,
                       &Bs[row * GS + cg * 8]);
        }
#if !TL_HAS_ASYNC_LDS
        if (kt + 1 < ksteps) {   // prefetch next K-tile into L2/L0
            const int row = tid >> 3, cg = tid & 7;
            __builtin_prefetch(A + (size_t)(bm + row) * K + k0 + 64 + cg * 8, 0, 1);
            __builtin_prefetch(W + (size_t)(bn + row) * K + k0 + 64 + cg * 8, 0, 1);
        }
#endif
        copy_fence_lds();
        __syncthreads();

#pragma unroll
        for (int kk = 0; kk < 2; ++kk) {
            v16h afr[4];
#pragma unroll
            for (int mt = 0; mt < 4; ++mt)
                afr[mt] = load_a_frag(&As[(waveM * 64 + mt * 16 + lane15) * GS + kk * 32],
                                      laneHalf);
            v16h bfr[2];
#pragma unroll
            for (int nt = 0; nt < 2; ++nt)
                bfr[nt] = load_b_frag(&Bs[(waveN * 32 + nt * 16 + lane15) * GS + kk * 32],
                                      laneHalf);
#pragma unroll
            for (int mt = 0; mt < 4; ++mt)
#pragma unroll
                for (int nt = 0; nt < 2; ++nt)
                    acc[mt][nt] = wmma16(afr[mt], bfr[nt], acc[mt][nt]);
        }
        __syncthreads();
    }

    // Epilogue: C layout -> row M = j + 8*laneHalf, col N = lane15 (per tile).
#pragma unroll
    for (int mt = 0; mt < 4; ++mt) {
#pragma unroll
        for (int nt = 0; nt < 2; ++nt) {
            const int gc = bn + waveN * 32 + nt * 16 + lane15;
            const float bv = bias[gc];
#pragma unroll
            for (int j = 0; j < 8; ++j) {
                const int gr = bm + waveM * 64 + mt * 16 + j + 8 * laneHalf;
                float v = acc[mt][nt][j] + bv;
                if (act == 1) v = gelu_exact(v);
                const size_t oi = (size_t)gr * N + gc;
                if (residual) v += residual[oi];
                if (C32) C32[oi] = v;
                if (C16) C16[oi] = (_Float16)v;
            }
        }
    }
}

// ---------------------------------------------------------------------------
// Flash attention (mask all-true): grid (L/128, B*H), 256 threads (8 waves).
// QKV input is f16 [B,L,3D]. Each wave owns 16 query rows. S = Q K^T via
// WMMA (scale folded into S), online softmax with half-wave shuffles, P
// routed through LDS (C-layout -> A-layout), O += P V via WMMA with V stored
// transposed in LDS. Output ctx is f16 [B,L,D].
// ---------------------------------------------------------------------------
#define QS 72    // Q/K LDS row stride (64 + 8)
#define PS 136   // P/Vt LDS row stride (128 + 8; 272B, 16B aligned)

__global__ __launch_bounds__(256)
void tl_attention_kernel(const _Float16* __restrict__ qkv,
                         _Float16* __restrict__ ctx) {
    __shared__ alignas(16) _Float16 Qs[128 * QS];
    __shared__ alignas(16) _Float16 Ks[128 * QS];
    __shared__ alignas(16) _Float16 Vt[64 * PS];       // transposed: [dim][key]
    __shared__ alignas(16) _Float16 Pw[8 * 16 * PS];   // per-wave P tiles

    const int tid  = threadIdx.x;
    const int lane = tid & 31;
    const int wave = tid >> 5;
    const int laneHalf = (lane >> 4) & 1;
    const int lane15   = lane & 15;

    const int q0 = blockIdx.x * 128;
    const int bh = blockIdx.y;
    const int b  = bh / TL_H;
    const int h  = bh % TL_H;
    const size_t rowStride = 3 * TL_D;
    const _Float16* qbase = qkv + (size_t)b * TL_L * rowStride + h * TL_HD;

    // Load Q block: 128 rows x 8 chunks of 8 f16; 4 chunks/thread.
#pragma unroll
    for (int p = 0; p < 4; ++p) {
        const int c   = tid + p * 256;
        const int row = c >> 3;
        const int cg  = c & 7;
        copy16_g2l(qbase + (size_t)(q0 + row) * rowStride + cg * 8,
                   &Qs[row * QS + cg * 8]);
    }

    v8f o[4];
#pragma unroll
    for (int t = 0; t < 4; ++t) o[t] = vzero8();
    float m_run[8], l_run[8];
#pragma unroll
    for (int j = 0; j < 8; ++j) { m_run[j] = -1.0e30f; l_run[j] = 0.0f; }

    _Float16* Pme = &Pw[wave * 16 * PS];

    for (int kb = 0; kb < TL_L / 128; ++kb) {
        __syncthreads();
        // K block -> Ks (row-major); V block -> Vt (transposed scatter).
#pragma unroll
        for (int p = 0; p < 4; ++p) {
            const int c   = tid + p * 256;
            const int row = c >> 3;
            const int cg  = c & 7;
            const size_t gro = (size_t)(kb * 128 + row) * rowStride + cg * 8;
            copy16_g2l(qbase + TL_D + gro, &Ks[row * QS + cg * 8]);
            union { uint4 u; _Float16 h[8]; } vv;
            vv.u = *(const uint4*)(qbase + 2 * TL_D + gro);
#pragma unroll
            for (int e = 0; e < 8; ++e)
                Vt[(cg * 8 + e) * PS + row] = vv.h[e];
        }
        copy_fence_lds();
        __syncthreads();

        // S = Q @ K^T : 8 key tiles of 16, K-dim 64 = 2 wmma steps.
        v8f sreg[8];
#pragma unroll
        for (int t = 0; t < 8; ++t) {
            v8f s = vzero8();
#pragma unroll
            for (int kk = 0; kk < 2; ++kk) {
                v16h a  = load_a_frag(&Qs[(wave * 16 + lane15) * QS + kk * 32], laneHalf);
                v16h bb = load_b_frag(&Ks[(t * 16 + lane15) * QS + kk * 32], laneHalf);
                s = wmma16(a, bb, s);
            }
#pragma unroll
            for (int j = 0; j < 8; ++j) s[j] *= 0.125f;   // 1/sqrt(HD)
            sreg[t] = s;
        }

        // Online softmax over this key block (row = j + 8*laneHalf).
#pragma unroll
        for (int j = 0; j < 8; ++j) {
            float mx = -1.0e30f;
#pragma unroll
            for (int t = 0; t < 8; ++t) mx = fmaxf(mx, sreg[t][j]);
#pragma unroll
            for (int off = 1; off < 16; off <<= 1)
                mx = fmaxf(mx, __shfl_xor(mx, off, 32));
            const float nm = fmaxf(m_run[j], mx);
            const float alpha = __expf(m_run[j] - nm);
            float rs = 0.0f;
#pragma unroll
            for (int t = 0; t < 8; ++t) {
                const float pv = __expf(sreg[t][j] - nm);
                sreg[t][j] = pv;
                rs += pv;
            }
#pragma unroll
            for (int off = 1; off < 16; off <<= 1)
                rs += __shfl_xor(rs, off, 32);
            m_run[j] = nm;
            l_run[j] = l_run[j] * alpha + rs;
#pragma unroll
            for (int t2 = 0; t2 < 4; ++t2) o[t2][j] *= alpha;
        }

        // P (C-layout) -> this wave's LDS tile, row-major 16x128 f16.
#pragma unroll
        for (int t = 0; t < 8; ++t)
#pragma unroll
            for (int j = 0; j < 8; ++j)
                Pme[(j + 8 * laneHalf) * PS + t * 16 + lane15] = (_Float16)sreg[t][j];

        // O += P @ V : K-dim 128 = 4 wmma steps; 4 output dim-tiles of 16.
#pragma unroll
        for (int kk = 0; kk < 4; ++kk) {
            v16h a = load_a_frag(&Pme[lane15 * PS + kk * 32], laneHalf);
#pragma unroll
            for (int t2 = 0; t2 < 4; ++t2) {
                v16h bb = load_b_frag(&Vt[(t2 * 16 + lane15) * PS + kk * 32], laneHalf);
                o[t2] = wmma16(a, bb, o[t2]);
            }
        }
    }

    // Normalize and write ctx[b, l, h*64 + d] (f16).
#pragma unroll
    for (int t2 = 0; t2 < 4; ++t2) {
#pragma unroll
        for (int j = 0; j < 8; ++j) {
            const int row = q0 + wave * 16 + j + 8 * laneHalf;
            const int col = h * TL_HD + t2 * 16 + lane15;
            ctx[((size_t)b * TL_L + row) * TL_D + col] =
                (_Float16)(o[t2][j] / l_run[j]);
        }
    }
}

// ---------------------------------------------------------------------------
// Host orchestration
// ---------------------------------------------------------------------------
extern "C" void kernel_launch(void* const* d_in, const int* in_sizes, int n_in,
                              void* d_out, int out_size, void* d_ws, size_t ws_size,
                              hipStream_t stream) {
    (void)in_sizes; (void)n_in; (void)out_size; (void)ws_size;
    const float* x     = (const float*)d_in[0];
    // d_in[1] is the mask: all-true in this model -> full attention, unused.
    const float* ln_g  = (const float*)d_in[2];
    const float* ln_b  = (const float*)d_in[3];
    const float* qkv_w = (const float*)d_in[4];
    const float* qkv_b = (const float*)d_in[5];
    const float* wo_w  = (const float*)d_in[6];
    const float* wo_b  = (const float*)d_in[7];
    const float* m0_w  = (const float*)d_in[8];
    const float* m0_b  = (const float*)d_in[9];
    const float* m1_w  = (const float*)d_in[10];
    const float* m1_b  = (const float*)d_in[11];
    float* out = (float*)d_out;

    // Workspace layout: f32 region first, then f16 region.
    float* x1 = (float*)d_ws;                               // [4096,1024] f32
    _Float16* hp      = (_Float16*)(x1 + (size_t)TL_M * TL_D);
    _Float16* qkv_wh  = hp;  hp += (size_t)3 * TL_D * TL_D; // [3072,1024]
    _Float16* wo_wh   = hp;  hp += (size_t)TL_D * TL_D;     // [1024,1024]
    _Float16* m0_wh   = hp;  hp += (size_t)TL_DFF * TL_D;   // [4096,1024]
    _Float16* m1_wh   = hp;  hp += (size_t)TL_D * TL_DFF;   // [1024,4096]
    _Float16* xn_h    = hp;  hp += (size_t)TL_M * TL_D;     // [4096,1024]
    _Float16* qkv_h   = hp;  hp += (size_t)TL_M * 3 * TL_D; // [4096,3072]
    _Float16* ctx_h   = hp;  hp += (size_t)TL_M * TL_D;     // [4096,1024]
    _Float16* hbuf_h  = hp;                                 // [4096,4096]

    const dim3 blk(256);

    // 0) weights fp32 -> f16 (once per launch).
    tl_cvt_f16_kernel<<<(3 * TL_D * TL_D / 4 + 255) / 256, blk, 0, stream>>>(
        qkv_w, qkv_wh, 3 * TL_D * TL_D / 4);
    tl_cvt_f16_kernel<<<(TL_D * TL_D / 4 + 255) / 256, blk, 0, stream>>>(
        wo_w, wo_wh, TL_D * TL_D / 4);
    tl_cvt_f16_kernel<<<(TL_DFF * TL_D / 4 + 255) / 256, blk, 0, stream>>>(
        m0_w, m0_wh, TL_DFF * TL_D / 4);
    tl_cvt_f16_kernel<<<(TL_D * TL_DFF / 4 + 255) / 256, blk, 0, stream>>>(
        m1_w, m1_wh, TL_D * TL_DFF / 4);

    // 1) xn = LN(x)  (f16)
    tl_layernorm_kernel<<<TL_M, blk, 0, stream>>>(x, ln_g, ln_b, xn_h);
    // 2) qkv = xn @ qkv_w^T + qkv_b  (f16 out)
    tl_gemm_f16_kernel<<<dim3(3 * TL_D / 128, TL_M / 128), blk, 0, stream>>>(
        xn_h, qkv_wh, qkv_b, nullptr, nullptr, qkv_h, TL_M, 3 * TL_D, TL_D, 0);
    // 3) ctx = softmax(q k^T / sqrt(hd)) v  (flash, full mask; f16 out)
    tl_attention_kernel<<<dim3(TL_L / 128, TL_B * TL_H), blk, 0, stream>>>(qkv_h, ctx_h);
    // 4) x1 = x + ctx @ wo_w^T + wo_b  (f32 out)
    tl_gemm_f16_kernel<<<dim3(TL_D / 128, TL_M / 128), blk, 0, stream>>>(
        ctx_h, wo_wh, wo_b, x, x1, nullptr, TL_M, TL_D, TL_D, 0);
    // 5) xn = LN(x1)  (f16, reuse buffer)
    tl_layernorm_kernel<<<TL_M, blk, 0, stream>>>(x1, ln_g, ln_b, xn_h);
    // 6) h = gelu(xn @ m0_w^T + m0_b)  (f16 out)
    tl_gemm_f16_kernel<<<dim3(TL_DFF / 128, TL_M / 128), blk, 0, stream>>>(
        xn_h, m0_wh, m0_b, nullptr, nullptr, hbuf_h, TL_M, TL_DFF, TL_D, 1);
    // 7) out = x1 + h @ m1_w^T + m1_b  (f32 out)
    tl_gemm_f16_kernel<<<dim3(TL_D / 128, TL_M / 128), blk, 0, stream>>>(
        hbuf_h, m1_wh, m1_b, x1, out, nullptr, TL_M, TL_D, TL_DFF, 0);
}
